// Bezier2Image_42777874268258
// MI455X (gfx1250) — matrix operand order repaired and checked by reading, not compiled, not run on hardware
//
#include <hip/hip_runtime.h>

// Bezier splat -> image as a per-batch 64x64x4800 f16 WMMA GEMM.
// res[b,w,v] = sum_k exp(-(bX[w]-X[k])^2/a) * exp(-(bY[v]-Y[k])^2/a), clamp to 1.

typedef __attribute__((ext_vector_type(16))) _Float16    v16h;
typedef __attribute__((ext_vector_type(8)))  float       v8f;
typedef __attribute__((ext_vector_type(4)))  float       v4f;
typedef __attribute__((ext_vector_type(4)))  unsigned int v4u;

#define LCUR   160
#define NSAMP  30
#define WPIX   60
#define KTOT   (LCUR * NSAMP)   // 4800
#define NCHUNK (KTOT / 32)      // 150
#define INV_ALPHA (1.0f / 2e-4f)

__device__ __forceinline__ unsigned int packh2(float a, float b) {
    _Float16 ha = (_Float16)a, hb = (_Float16)b;
    unsigned short ua, ub;
    __builtin_memcpy(&ua, &ha, 2);
    __builtin_memcpy(&ub, &hb, 2);
    return (unsigned int)ua | ((unsigned int)ub << 16);
}

__global__ __launch_bounds__(256)
void bezier_splat_wmma(const float* __restrict__ xin, float* __restrict__ out)
{
    // Sample coordinates for this batch
    __shared__ __attribute__((aligned(16))) float Xs[KTOT];
    __shared__ __attribute__((aligned(16))) float Ys[KTOT];
    // Double-buffered WMMA-fragment-layout tiles: [buf][tile][lane][8 dwords]
    __shared__ __attribute__((aligned(32))) unsigned int Atile[2][4][32][8]; // gX, A 16x32 layout
    __shared__ __attribute__((aligned(32))) unsigned int Btile[2][4][32][8]; // gY, B 32x16 layout

    const int tid = threadIdx.x;
    const int b   = blockIdx.x;

    // ---------- Phase 1: evaluate 4800 bezier sample points into LDS ----------
    const float* xb = xin + (size_t)b * (LCUR * 8);
    for (int p = tid; p < KTOT; p += 256) {
        int l  = p / NSAMP;
        int ns = p - l * NSAMP;
        float t = (float)ns * (1.0f / (float)NSAMP);
        t = 2.0f * t * t * t - 3.0f * t * t + 2.0f * t;   // reparametrize
        float t2 = t * t, t3 = t2 * t;
        float T0 = t3;
        float T1 = 3.0f * (t2 - t3);
        float T2 = 3.0f * (t3 - 2.0f * t2 + t);
        float omt = 1.0f - t;
        float T3 = omt * omt * omt;
        const float* cp = xb + l * 8;  // [4 ctrl pts][2 coords]
        Xs[p] = T0 * cp[0] + T1 * cp[2] + T2 * cp[4] + T3 * cp[6];
        Ys[p] = T0 * cp[1] + T1 * cp[3] + T2 * cp[5] + T3 * cp[7];
    }
    __syncthreads();

    // ---------- Wave -> output-tile assignment (16 tiles, 8 waves, 2 each) ----------
    const int lane = tid & 31;
    const int wv   = tid >> 5;
    const int mt   = wv >> 1;            // A tile row   (w block)
    const int nt0  = (wv & 1) * 2;       // B tile col 0 (v block)
    const int nt1  = nt0 + 1;

    v8f acc0 = {};
    v8f acc1 = {};

    // ---------- Fill-role indices (waves 0-3 build A/gX, waves 4-7 build B/gY) ----------
    const bool fillA = (tid < 128);
    const int  ft     = tid & 127;
    const int  f_tile = ft >> 5;          // 0..3
    const int  f_row  = (ft >> 1) & 15;   // m (A) or n (B), 0..15
    const int  f_kh   = ft & 1;           // which 16-wide K half of the 32-chunk
    const float pixc  = (float)(16 * f_tile + f_row) * (1.0f / (float)WPIX);
    const float* S    = fillA ? Xs : Ys;

    for (int ck = 0; ck < NCHUNK; ++ck) {
        const int buf = ck & 1;
        const int k0  = ck * 32 + f_kh * 16;

        // 16 gaussians for this thread's (row, K-half)
        float g[16];
        #pragma unroll
        for (int q = 0; q < 4; ++q) {
            v4f sv = *(const v4f*)&S[k0 + 4 * q];
            #pragma unroll
            for (int i = 0; i < 4; ++i) {
                float d = pixc - sv[i];
                g[4 * q + i] = __expf(-(d * d) * INV_ALPHA);
            }
        }
        v4u lo = { packh2(g[0], g[1]),  packh2(g[2], g[3]),
                   packh2(g[4], g[5]),  packh2(g[6], g[7]) };
        v4u hi = { packh2(g[8], g[9]),  packh2(g[10], g[11]),
                   packh2(g[12], g[13]), packh2(g[14], g[15]) };

        if (fillA) {
            // A 16x32 f16 layout: kk=f_kh*16+i.
            //   i=0..7  -> lane = m,    dwords f_kh*4 .. +3
            //   i=8..15 -> lane = m+16, dwords f_kh*4 .. +3
            *(v4u*)&Atile[buf][f_tile][f_row     ][f_kh * 4] = lo;
            *(v4u*)&Atile[buf][f_tile][f_row + 16][f_kh * 4] = hi;
        } else {
            // B 32x16 f16 layout: kk=f_kh*16+i; i=0..7 -> lane = n, i=8..15 -> lane = n+16, dwords f_kh*4 .. +3
            *(v4u*)&Btile[buf][f_tile][f_row     ][f_kh * 4] = lo;
            *(v4u*)&Btile[buf][f_tile][f_row + 16][f_kh * 4] = hi;
        }

        __syncthreads();  // double-buffered: one barrier per chunk

        v16h a  = *(const v16h*)&Atile[buf][mt ][lane][0];
        v16h b0 = *(const v16h*)&Btile[buf][nt0][lane][0];
        v16h b1 = *(const v16h*)&Btile[buf][nt1][lane][0];
        acc0 = __builtin_amdgcn_wmma_f32_16x16x32_f16(false, a, false, b0,
                                                      (short)0, acc0, false, false);
        acc1 = __builtin_amdgcn_wmma_f32_16x16x32_f16(false, a, false, b1,
                                                      (short)0, acc1, false, false);
    }

    // ---------- Epilogue: C/D layout -> out[b][w][v], clamp to 1 ----------
    const int h    = lane >> 4;
    const int ncol = lane & 15;
    float* ob = out + (size_t)b * (WPIX * WPIX);
    #pragma unroll
    for (int i = 0; i < 8; ++i) {
        int w = 16 * mt + 8 * h + i;
        if (w < WPIX) {
            int v0 = 16 * nt0 + ncol;
            int v1 = 16 * nt1 + ncol;
            if (v0 < WPIX) ob[w * WPIX + v0] = fminf(acc0[i], 1.0f);
            if (v1 < WPIX) ob[w * WPIX + v1] = fminf(acc1[i], 1.0f);
        }
    }
}

extern "C" void kernel_launch(void* const* d_in, const int* in_sizes, int n_in,
                              void* d_out, int out_size, void* d_ws, size_t ws_size,
                              hipStream_t stream) {
    const float* x = (const float*)d_in[0];
    float* out = (float*)d_out;
    const int B = in_sizes[0] / (LCUR * 8);   // 128
    bezier_splat_wmma<<<dim3(B), dim3(256), 0, stream>>>(x, out);
}
